// Block_8933531976055
// MI455X (gfx1250) — matrix operand order repaired and checked by reading, not compiled, not run on hardware
//
#include <hip/hip_runtime.h>
#include <hip/hip_bf16.h>

// MI455X / gfx1250 graph-ODE: h = x@Wfc+b; 4x Euler steps of two fused
// GCN layers. Aggregation adj@h with V_WMMA_F32_16X16X32_BF16 fed by
// GLOBAL_LOAD_ASYNC_TO_LDS_B128 double-buffered tiles; projection @W
// fused in the same kernel via LDS restaging.

typedef __bf16 bf16_t;
typedef __attribute__((ext_vector_type(8)))  __bf16 v8bf;
typedef __attribute__((ext_vector_type(16))) __bf16 v16bf;
typedef __attribute__((ext_vector_type(8)))  float  v8f;
typedef __attribute__((ext_vector_type(4)))  float  v4f;

#define NNODE 4096
#define HDIM  128
#define BATCH 32
#define INDIM 64
#define KT    (NNODE / 32)   // 128 K-tiles of depth 32

static __device__ __forceinline__ v16bf cat16(v8bf lo, v8bf hi) {
  v16bf r;
#pragma unroll
  for (int e = 0; e < 8; ++e) { r[e] = lo[e]; r[e + 8] = hi[e]; }
  return r;
}

// Async DMA of one 16-byte chunk: global (saddr base + 32-bit voffset) -> LDS.
// Tracked by ASYNCcnt, no VGPR staging.
static __device__ __forceinline__ void async_ld16(unsigned lds_addr,
                                                  const bf16_t* base,
                                                  unsigned goff) {
  asm volatile("global_load_async_to_lds_b128 %0, %1, %2"
               :: "v"(lds_addr), "v"(goff), "s"(base)
               : "memory");
}

// ---------------------------------------------------------------- adj -> bf16
__global__ __launch_bounds__(256) void cvt_adj(const float* __restrict__ a,
                                               v8bf* __restrict__ out, int n8) {
  int g = blockIdx.x * blockDim.x + threadIdx.x;
  if (g >= n8) return;
  const v4f* src = (const v4f*)a;
  v4f f0 = src[2 * g], f1 = src[2 * g + 1];
  v8bf o;
#pragma unroll
  for (int e = 0; e < 4; ++e) { o[e] = (__bf16)f0[e]; o[e + 4] = (__bf16)f1[e]; }
  out[g] = o;
}

// ----------------------------------------------------- h0 = x @ Wfc + bfc
// writes h_f32 [B][N][H] and transposed bf16 mirror hT [B][H][N]
__global__ __launch_bounds__(256) void fc_kernel(const float* __restrict__ x,
                                                 const float* __restrict__ Wfc,
                                                 const float* __restrict__ bfc,
                                                 float* __restrict__ hf,
                                                 bf16_t* __restrict__ hT) {
  __shared__ float sx[8 * INDIM];
  const int t = threadIdx.x;
  const long bn0 = (long)blockIdx.x * 8;      // flat row index over B*N
  sx[t]       = x[bn0 * INDIM + t];
  sx[t + 256] = x[bn0 * INDIM + t + 256];
  __syncthreads();
  const int col = t & 127;
  const int r0  = t >> 7;
  for (int rr = r0; rr < 8; rr += 2) {
    float sum = bfc[col];
#pragma unroll
    for (int k = 0; k < INDIM; ++k)
      sum = fmaf(sx[rr * INDIM + k], Wfc[k * HDIM + col], sum);
    long flat = bn0 + rr;                     // node index over B*N
    long b  = flat >> 12;
    long nn = flat & (NNODE - 1);
    hf[flat * HDIM + col] = sum;
    hT[(b * HDIM + col) * NNODE + nn] = (__bf16)sum;
  }
}

// ------------------------------------------------ fused GCN layer (WMMA)
// out = relu( (adj @ h) @ W + bias ); mode 1 additionally does the Euler
// residual h_f32 += 0.25*out and writes the bf16 transposed mirror.
// Workgroup tile: 256 nodes x 128 hidden; each of 8 waves owns 32 rows
// (2 A-fragments, 16 f32 accumulators).
__global__ __launch_bounds__(256)
void gcn_fused(const bf16_t* __restrict__ adjb,   // [N][N] bf16 row-major
               const bf16_t* __restrict__ bT,     // [B][H][N] bf16 (transposed B operand)
               const float*  __restrict__ Wf,     // [H][H] f32
               const float*  __restrict__ biasf,  // [H]
               float*        __restrict__ hf,     // [B][N][H] f32 (mode 1 only)
               bf16_t*       __restrict__ outT,   // [B][H][N] bf16 output
               int mode) {
  // 96 KB shared, manually unioned (all accesses via one typed AS(3) array):
  //   main loop : bufA0 @vec 0 (1024)  bufB0 @vec 1024 (512)
  //               bufA1 @vec 1536      bufB1 @vec 2560   (48 KB total)
  //   stage 2   : sAcc  @vec 0 (4096 = 256x128 bf16)
  //               sW    @vec 4096 (2048 = 128x128 bf16)
  __shared__ v8bf smem[6144];

  const int t    = threadIdx.x;
  const int lane = t & 31;
  const int wave = t >> 5;
  const int m0   = blockIdx.x * 256;
  const long b   = blockIdx.y;
  const int halfsel = (lane < 16) ? 0 : 1;
  const int lrow    = lane & 15;

  const unsigned smemBase = (unsigned)(uintptr_t)&smem[0];

  // Per-thread DMA chunk descriptors (16 B each).
  // A tile: 256 rows x 32 K  -> 1024 chunks, 4/thread
  // B tile: 128 hcol x 32 K  ->  512 chunks, 2/thread
  unsigned goffA[4], ldsA[4], goffB[2], ldsB[2];
#pragma unroll
  for (int i = 0; i < 4; ++i) {
    int c = t + 256 * i, r = c >> 2, q = c & 3;
    goffA[i] = (unsigned)(((m0 + r) * NNODE + 8 * q) * 2);
    ldsA[i]  = (unsigned)((r * 4 + q) * 16);
  }
#pragma unroll
  for (int i = 0; i < 2; ++i) {
    int c = t + 256 * i, r = c >> 2, q = c & 3;
    goffB[i] = (unsigned)((((int)b * HDIM + r) * NNODE + 8 * q) * 2);
    ldsB[i]  = (unsigned)((r * 4 + q) * 16);
  }

  auto issueTile = [&](int kt, int bufsel) {
    unsigned kb    = (unsigned)(kt * 64);                  // K-offset in bytes
    unsigned aBase = smemBase + (bufsel ? 24576u : 0u);
    unsigned bBase = smemBase + (bufsel ? 40960u : 16384u);
#pragma unroll
    for (int i = 0; i < 4; ++i) async_ld16(aBase + ldsA[i], adjb, goffA[i] + kb);
#pragma unroll
    for (int i = 0; i < 2; ++i) async_ld16(bBase + ldsB[i], bT,   goffB[i] + kb);
  };

  v8f acc[16];
#pragma unroll
  for (int j = 0; j < 16; ++j)
#pragma unroll
    for (int i = 0; i < 8; ++i) acc[j][i] = 0.0f;

  issueTile(0, 0);

  const int r0 = wave * 32 + lrow;
  for (int kt = 0; kt < KT; ++kt) {
    const int cur = kt & 1;
    if (kt + 1 < KT) {
      issueTile(kt + 1, cur ^ 1);
      // 6 just-issued chunks may stay outstanding; async loads complete
      // in order, so <=6 guarantees the current tile is resident.
      asm volatile("s_wait_asynccnt 6" ::: "memory");
    } else {
      asm volatile("s_wait_asynccnt 0" ::: "memory");
    }
    __syncthreads();

    const int aOff = cur ? 1536 : 0;
    const int bOff = cur ? 2560 : 1024;
    // A frags: lane<16 holds K{0..7,16..23}, lane>=16 K{8..15,24..31}
    v16bf a0 = cat16(smem[aOff + r0 * 4 + halfsel],
                     smem[aOff + r0 * 4 + 2 + halfsel]);
    v16bf a1 = cat16(smem[aOff + (r0 + 16) * 4 + halfsel],
                     smem[aOff + (r0 + 16) * 4 + 2 + halfsel]);
#pragma unroll
    for (int j = 0; j < 8; ++j) {
      int ncol = j * 16 + lrow;
      // B frag: lane<16 K{0..7,16..23}, lane>=16 K{8..15,24..31}
      v16bf bf = cat16(smem[bOff + ncol * 4 + halfsel],
                       smem[bOff + ncol * 4 + 2 + halfsel]);
      acc[j]     = __builtin_amdgcn_wmma_f32_16x16x32_bf16(
          false, a0, false, bf, (short)0, acc[j], false, false);
      acc[8 + j] = __builtin_amdgcn_wmma_f32_16x16x32_bf16(
          false, a1, false, bf, (short)0, acc[8 + j], false, false);
    }
    __syncthreads();
  }

  // ---- stage 2: projection relu(acc @ W + bias), fused in-WG ----
  bf16_t* sAcc = (bf16_t*)&smem[0];     // [256 node][128 hin] bf16
  bf16_t* sW   = (bf16_t*)&smem[4096];  // [128 hout][128 hin] bf16 (W^T)
#pragma unroll
  for (int s2 = 0; s2 < 2; ++s2)
#pragma unroll
    for (int j = 0; j < 8; ++j) {
      int coln = j * 16 + lrow;
#pragma unroll
      for (int i = 0; i < 8; ++i) {
        int rown = wave * 32 + s2 * 16 + ((lane < 16) ? i : i + 8);
        sAcc[rown * HDIM + coln] = (__bf16)acc[s2 * 8 + j][i];
      }
    }
  for (int it = 0; it < 16; ++it) {
    int c  = t + 256 * it;         // float4 chunk id, [0,4096)
    int k  = c >> 5;               // hin
    int n4 = (c & 31) * 4;         // hout base
    v4f w = *(const v4f*)(Wf + k * HDIM + n4);
#pragma unroll
    for (int e = 0; e < 4; ++e) sW[(n4 + e) * HDIM + k] = (__bf16)w[e];
  }
  __syncthreads();

  // reuse acc registers for the projection accumulators
#pragma unroll
  for (int j = 0; j < 16; ++j)
#pragma unroll
    for (int i = 0; i < 8; ++i) acc[j][i] = 0.0f;

#pragma unroll
  for (int k2 = 0; k2 < 4; ++k2) {
#pragma unroll
    for (int s2 = 0; s2 < 2; ++s2) {
      int arow = wave * 32 + s2 * 16 + lrow;
      v16bf a2 = cat16(smem[arow * 16 + k2 * 4 + halfsel],
                       smem[arow * 16 + k2 * 4 + 2 + halfsel]);
#pragma unroll
      for (int j = 0; j < 8; ++j) {
        int ncol = j * 16 + lrow;
        v16bf b2 = cat16(smem[4096 + ncol * 16 + k2 * 4 + halfsel],
                         smem[4096 + ncol * 16 + k2 * 4 + 2 + halfsel]);
        acc[s2 * 8 + j] = __builtin_amdgcn_wmma_f32_16x16x32_bf16(
            false, a2, false, b2, (short)0, acc[s2 * 8 + j], false, false);
      }
    }
  }

  // ---- epilogue: bias + relu (+ Euler residual in mode 1) ----
#pragma unroll
  for (int s2 = 0; s2 < 2; ++s2)
#pragma unroll
    for (int j = 0; j < 8; ++j) {
      int n = j * 16 + lrow;                       // hidden index
      float bv = biasf[n];
#pragma unroll
      for (int i = 0; i < 8; ++i) {
        int row = m0 + wave * 32 + s2 * 16 + ((lane < 16) ? i : i + 8);
        float v = acc[s2 * 8 + j][i] + bv;
        v = v > 0.0f ? v : 0.0f;
        long oT = (b * HDIM + n) * NNODE + row;
        if (mode == 0) {
          outT[oT] = (__bf16)v;                    // t1 (bf16 mirror only)
        } else {
          long of = (b * NNODE + row) * HDIM + n;
          float hn = hf[of] + 0.25f * v;           // Euler step
          hf[of] = hn;
          outT[oT] = (__bf16)hn;                   // refresh bf16 mirror
        }
      }
    }
}

extern "C" void kernel_launch(void* const* d_in, const int* in_sizes, int n_in,
                              void* d_out, int out_size, void* d_ws, size_t ws_size,
                              hipStream_t stream) {
  const float* x   = (const float*)d_in[0];
  const float* adj = (const float*)d_in[1];
  const float* Wfc = (const float*)d_in[2];
  const float* bfc = (const float*)d_in[3];
  const float* W1  = (const float*)d_in[4];
  const float* b1  = (const float*)d_in[5];
  const float* W2  = (const float*)d_in[6];
  const float* b2  = (const float*)d_in[7];

  char* ws = (char*)d_ws;
  bf16_t* adjb = (bf16_t*)ws; ws += (size_t)NNODE * NNODE * 2;            // 32 MB
  float*  hf   = (float*)ws;  ws += (size_t)BATCH * NNODE * HDIM * 4;     // 64 MB
  bf16_t* hT   = (bf16_t*)ws; ws += (size_t)BATCH * NNODE * HDIM * 2;     // 32 MB
  bf16_t* t1T  = (bf16_t*)ws;                                             // 32 MB

  // 1) adj -> bf16 (stays hot in the 192 MB L2 for all 8 GEMM passes)
  const int n8 = NNODE * NNODE / 8;
  cvt_adj<<<n8 / 256, 256, 0, stream>>>(adj, (v8bf*)adjb, n8);

  // 2) encoder: h0 = x @ Wfc + bfc
  fc_kernel<<<(BATCH * NNODE) / 8, 256, 0, stream>>>(x, Wfc, bfc, hf, hT);

  // 3) 4 Euler steps; two fused WMMA GCN layers per step
  dim3 grid(NNODE / 256, BATCH);
  for (int s = 0; s < 4; ++s) {
    gcn_fused<<<grid, 256, 0, stream>>>(adjb, hT,  W1, b1, nullptr, t1T, 0);
    gcn_fused<<<grid, 256, 0, stream>>>(adjb, t1T, W2, b2, hf,      hT,  1);
  }

  // 4) final h (fp32) -> output
  hipMemcpyAsync(d_out, hf, (size_t)BATCH * NNODE * HDIM * 4,
                 hipMemcpyDeviceToDevice, stream);
}
